// MLASegmentCrossAttention_37469294690562
// MI455X (gfx1250) — matrix operand order repaired and checked by reading, not compiled, not run on hardware
//
#include <hip/hip_runtime.h>
#include <hip/hip_bf16.h>

// ---------------- problem constants ----------------
constexpr int BB   = 4;
constexpr int LQ   = 2048;
constexpr int LK   = 512;
constexpr int QDIM = 1024;
constexpr int KVDIM= 1024;
constexpr int H    = 16;
constexpr int HD   = 64;
constexpr int KVC  = 256;
constexpr int QC   = 384;
constexpr int RR   = 64;

constexpr int NTQ = BB * LQ;   // 8192 query tokens
constexpr int NTK = BB * LK;   // 2048 kv tokens

typedef __bf16 bf16;
typedef __attribute__((ext_vector_type(16))) __bf16 v16bf;
typedef __attribute__((ext_vector_type(8)))  float  v8f;

__device__ __forceinline__ float bf2f(bf16 x) { return (float)x; }
__device__ __forceinline__ bf16  f2bf(float x){ return (bf16)x; }

// ---------------- fp32 -> bf16 convert ----------------
__global__ void cvt_f32_bf16(const float* __restrict__ in, bf16* __restrict__ out, long n) {
    long i = (long)blockIdx.x * blockDim.x + threadIdx.x;
    if (i < n) out[i] = f2bf(in[i]);
}

// fp32 [K][N] -> bf16 transposed [N][K]; coalesced writes, strided reads (weights only)
__global__ void cvt_f32_bf16_T(const float* __restrict__ in, bf16* __restrict__ out,
                               int Kd, int Nd) {
    long i = (long)blockIdx.x * blockDim.x + threadIdx.x;   // over N*K, output-major
    if (i >= (long)Kd * Nd) return;
    int n = (int)(i / Kd);
    int k = (int)(i % Kd);
    out[i] = f2bf(in[(size_t)k * Nd + n]);
}

// ---------------- async global -> LDS (CDNA5, ASYNCcnt-tracked) ----------------
// 16 bytes per call. LDS byte address = low 32 bits of the generic shared pointer.
__device__ __forceinline__ void async_b128(const bf16* __restrict__ g, unsigned lds) {
    unsigned long long ga = (unsigned long long)(uintptr_t)g;
    asm volatile("global_load_async_to_lds_b128 %0, %1, off"
                 :: "v"(lds), "v"(ga) : "memory");
}
__device__ __forceinline__ void wait_async0() {
    asm volatile("s_wait_asynccnt 0x0" ::: "memory");
}

// ---------------- WMMA fragment builders from LDS (ISA 7.12.2 layouts) ----------------
// A 16x32 bf16, LDS tile [rows][32]: lane = M row (xr), group g walks the
// 8/16-interleaved K pattern -> two contiguous 16B groups.
__device__ __forceinline__ v16bf frag_a(const bf16* As, int xr, int g) {
    const bf16* row = As + xr * 32;
    union { v16bf v; unsigned u[8]; } f;
#pragma unroll
    for (int j = 0; j < 4; ++j)                   // VGPR 0..3 : K = 8g + 2j, +1
        f.u[j] = *(const unsigned*)(row + 8 * g + 2 * j);
#pragma unroll
    for (int j = 0; j < 4; ++j)                   // VGPR 4..7 : K = 16 + 8g + 2j, +1
        f.u[4 + j] = *(const unsigned*)(row + 16 + 8 * g + 2 * j);
    return f.v;
}

// B 32x16 bf16 from TRANSPOSED weights: LDS tile Bt[n_local][32] row-major in K.
// Lane = N col (xr), same 8/16-interleaved K pattern as frag_a: K = 8g + s, 16 + 8g + (s-8).
__device__ __forceinline__ v16bf frag_bt(const bf16* Bs, int xr, int g) {
    const bf16* row = Bs + xr * 32 + 8 * g;
    union { v16bf v; unsigned u[8]; } f;
#pragma unroll
    for (int j = 0; j < 8; ++j)
        f.u[j] = *(const unsigned*)(row + 2 * j + (j >= 4 ? 8 : 0));
    return f.v;
}

// ---------------- typed store (no runtime branch in epilogue) ----------------
__device__ __forceinline__ void st_out(float* p, float v) { *p = v; }
__device__ __forceinline__ void st_out(bf16*  p, float v) { *p = f2bf(v); }

// ---------------- LDS-staged, double-buffered bf16 WMMA GEMM ----------------
// C[M,N] = A[M,K] x Bt[N,K]^T; block = 128 threads = 4 waves; block tile 64x64,
// wave tile 32x32; K stepped by 32 with async-LDS double buffering.
template <typename OutT>
__global__ __launch_bounds__(128)
void wmma_gemm(const bf16* __restrict__ A, const bf16* __restrict__ Bt,
               OutT* __restrict__ C, int M, int N, int K) {
    const int tid  = threadIdx.x;
    const int lane = tid & 31;
    const int wave = tid >> 5;
    const int wr   = wave >> 1;        // 0..1
    const int wc   = wave & 1;         // 0..1
    const int g    = lane >> 4;
    const int xr   = lane & 15;
    const int mBlk = blockIdx.y * 64;
    const int nBlk = blockIdx.x * 64;

    __shared__ __align__(16) bf16 As[2][64 * 32];   // 4KB per buffer
    __shared__ __align__(16) bf16 Bs[2][64 * 32];   // 4KB per buffer

    // staging geometry: 128 threads x (2 chunks/tile) x 16B cover each 64x32 tile
    const int srow = tid >> 2;                      // 0..31
    const int scol = (tid & 3) * 8;                 // 0,8,16,24
    const bf16* gA0 = A  + (size_t)(mBlk + srow)      * K + scol;
    const bf16* gA1 = A  + (size_t)(mBlk + srow + 32) * K + scol;
    const bf16* gB0 = Bt + (size_t)(nBlk + srow)      * K + scol;
    const bf16* gB1 = Bt + (size_t)(nBlk + srow + 32) * K + scol;
    const unsigned lA = (unsigned)(uintptr_t)&As[0][srow * 32 + scol];
    const unsigned lB = (unsigned)(uintptr_t)&Bs[0][srow * 32 + scol];
    // +32 rows => +2048 bytes; buffer 1 => +4096 bytes

    v8f c00 = {}; v8f c01 = {}; v8f c10 = {}; v8f c11 = {};

    // prologue: stage slab 0 into buffer 0
    async_b128(gA0, lA);
    async_b128(gA1, lA + 2048u);
    async_b128(gB0, lB);
    async_b128(gB1, lB + 2048u);

    int buf = 0;
    for (int k0 = 0; k0 < K; k0 += 32) {
        wait_async0();                 // my async copies into buffer `buf` done
        __syncthreads();               // everyone's copies visible; prev reads done

        if (k0 + 32 < K) {             // overlap next slab with this slab's WMMAs
            unsigned bo = (buf ^ 1) ? 4096u : 0u;
            int kn = k0 + 32;
            async_b128(gA0 + kn, lA + bo);
            async_b128(gA1 + kn, lA + bo + 2048u);
            async_b128(gB0 + kn, lB + bo);
            async_b128(gB1 + kn, lB + bo + 2048u);
        }

        const bf16* Aw = &As[buf][(size_t)(wr * 32) * 32];
        const bf16* Bw = &Bs[buf][(size_t)(wc * 32) * 32];
        v16bf a0 = frag_a (Aw,           xr, g);
        v16bf a1 = frag_a (Aw + 16 * 32, xr, g);
        v16bf b0 = frag_bt(Bw,           xr, g);
        v16bf b1 = frag_bt(Bw + 16 * 32, xr, g);

        c00 = __builtin_amdgcn_wmma_f32_16x16x32_bf16(false, a0, false, b0, (short)0, c00, false, false);
        c01 = __builtin_amdgcn_wmma_f32_16x16x32_bf16(false, a0, false, b1, (short)0, c01, false, false);
        c10 = __builtin_amdgcn_wmma_f32_16x16x32_bf16(false, a1, false, b0, (short)0, c10, false, false);
        c11 = __builtin_amdgcn_wmma_f32_16x16x32_bf16(false, a1, false, b1, (short)0, c11, false, false);

        buf ^= 1;
    }

    // C/D layout: VGPR r <-> m = r + 8*g, n = lane&15
    const int m0 = mBlk + wr * 32;
    const int n0 = nBlk + wc * 32;
#pragma unroll
    for (int r = 0; r < 8; ++r) {
        int mm = r + 8 * g;
        st_out(C + (size_t)(m0 + mm)      * N + n0 +      xr, c00[r]);
        st_out(C + (size_t)(m0 + mm)      * N + n0 + 16 + xr, c01[r]);
        st_out(C + (size_t)(m0 + 16 + mm) * N + n0 +      xr, c10[r]);
        st_out(C + (size_t)(m0 + 16 + mm) * N + n0 + 16 + xr, c11[r]);
    }
}

// ---------------- RoPE ----------------
__global__ void rope_q_kernel(bf16* __restrict__ x) {
    long idx = (long)blockIdx.x * blockDim.x + threadIdx.x;   // NTQ*H*32
    if (idx >= (long)NTQ * H * 32) return;
    int i = idx & 31;
    int h = (idx >> 5) & 15;
    int t = (int)(idx >> 9);
    int pos = t & (LQ - 1);
    float inv = powf(10000.0f, -(float)i / 32.0f);
    float ang = (float)pos * inv;
    float c = cosf(ang), s = sinf(ang);
    bf16* row = x + (size_t)t * (H * RR) + h * RR;
    float x1 = bf2f(row[i]), x2 = bf2f(row[32 + i]);
    row[i]      = f2bf(x1 * c - x2 * s);
    row[32 + i] = f2bf(x1 * s + x2 * c);
}

__global__ void rope_k_kernel(bf16* __restrict__ x) {
    long idx = (long)blockIdx.x * blockDim.x + threadIdx.x;   // NTK*32
    if (idx >= (long)NTK * 32) return;
    int i = idx & 31;
    int t = (int)(idx >> 5);
    int pos = t & (LK - 1);
    float inv = powf(10000.0f, -(float)i / 32.0f);
    float ang = (float)pos * inv;
    float c = cosf(ang), s = sinf(ang);
    bf16* row = x + (size_t)t * RR;
    float x1 = bf2f(row[i]), x2 = bf2f(row[32 + i]);
    row[i]      = f2bf(x1 * c - x2 * s);
    row[32 + i] = f2bf(x1 * s + x2 * c);
}

// ---------------- sparse 3-key attention (one wave per token,head) ----------------
__global__ __launch_bounds__(256)
void attn_sparse(const bf16* __restrict__ qn, const bf16* __restrict__ qr,
                 const bf16* __restrict__ kn, const bf16* __restrict__ kr,
                 const bf16* __restrict__ vv, const int* __restrict__ seg,
                 bf16* __restrict__ out) {
    const int lane = threadIdx.x & 31;
    const int wave = threadIdx.x >> 5;
    const long gid = (long)blockIdx.x * 8 + wave;   // NTQ*H total
    const int t = (int)(gid >> 4);                  // token (H=16)
    const int h = (int)(gid & 15);
    const int b = t >> 11;                          // LQ=2048
    const int s = seg[t];

    const int d0 = lane * 2;
    const bf16* qnr = qn + (size_t)t * (H * HD) + h * HD;
    const bf16* qrr = qr + (size_t)t * (H * RR) + h * RR;
    const float qn0 = bf2f(qnr[d0]), qn1 = bf2f(qnr[d0 + 1]);
    const float qr0 = bf2f(qrr[d0]), qr1 = bf2f(qrr[d0 + 1]);

    float sc[3]; int kk[3]; int nv = 0;
#pragma unroll
    for (int j = 2; j >= 0; --j) {
        int kj = s - j;
        if (kj < 0) continue;                       // uniform per wave
        size_t kb = (size_t)(b * LK + kj);
        const bf16* knr = kn + kb * (H * HD) + h * HD;
        const bf16* krr = kr + kb * RR;
        float p = qn0 * bf2f(knr[d0]) + qn1 * bf2f(knr[d0 + 1])
                + qr0 * bf2f(krr[d0]) + qr1 * bf2f(krr[d0 + 1]);
#pragma unroll
        for (int m = 16; m >= 1; m >>= 1) p += __shfl_xor(p, m, 32);
        sc[nv] = p * 0.0883883476483184f;           // 1/sqrt(HD+R)
        kk[nv] = kj;
        ++nv;
    }

    float mx = sc[0];
    for (int j = 1; j < nv; ++j) mx = fmaxf(mx, sc[j]);
    float e[3], den = 0.0f;
    for (int j = 0; j < nv; ++j) { e[j] = expf(sc[j] - mx); den += e[j]; }

    float o0 = 0.0f, o1 = 0.0f;
    for (int j = 0; j < nv; ++j) {
        float p = e[j] / den;
        const bf16* vr = vv + ((size_t)(b * LK + kk[j])) * (H * HD) + h * HD;
        o0 += p * bf2f(vr[d0]);
        o1 += p * bf2f(vr[d0 + 1]);
    }
    bf16* orow = out + (size_t)t * (H * HD) + h * HD;
    orow[d0]     = f2bf(o0);
    orow[d0 + 1] = f2bf(o1);
}

// ---------------- host-side orchestration ----------------
static inline void launch_cvt(const float* in, bf16* out, long n, hipStream_t s) {
    long blocks = (n + 255) / 256;
    cvt_f32_bf16<<<(unsigned)blocks, 256, 0, s>>>(in, out, n);
}
static inline void launch_cvtT(const float* in, bf16* out, int Kd, int Nd, hipStream_t s) {
    long n = (long)Kd * Nd;
    long blocks = (n + 255) / 256;
    cvt_f32_bf16_T<<<(unsigned)blocks, 256, 0, s>>>(in, out, Kd, Nd);
}
static inline void gemm_bf16out(const bf16* A, const bf16* Bt, bf16* C,
                                int M, int N, int K, hipStream_t s) {
    dim3 grid(N / 64, M / 64);
    wmma_gemm<bf16><<<grid, 128, 0, s>>>(A, Bt, C, M, N, K);
}
static inline void gemm_f32out(const bf16* A, const bf16* Bt, float* C,
                               int M, int N, int K, hipStream_t s) {
    dim3 grid(N / 64, M / 64);
    wmma_gemm<float><<<grid, 128, 0, s>>>(A, Bt, C, M, N, K);
}

extern "C" void kernel_launch(void* const* d_in, const int* in_sizes, int n_in,
                              void* d_out, int out_size, void* d_ws, size_t ws_size,
                              hipStream_t stream) {
    const float* q        = (const float*)d_in[0];
    const float* kv       = (const float*)d_in[1];
    const int*   seg_id   = (const int*)  d_in[2];
    const float* w_kvcomp = (const float*)d_in[3];
    const float* w_qdown  = (const float*)d_in[4];
    const float* w_qup    = (const float*)d_in[5];
    const float* w_qrope  = (const float*)d_in[6];
    const float* w_kup    = (const float*)d_in[7];
    const float* w_krope  = (const float*)d_in[8];
    const float* w_vup    = (const float*)d_in[9];
    const float* w_out    = (const float*)d_in[10];

    char* ws = (char*)d_ws;
    size_t off = 0;
    auto alloc_bf = [&](size_t elems) -> bf16* {
        bf16* p = (bf16*)(ws + off);
        off += ((elems * sizeof(bf16) + 255) / 256) * 256;
        return p;
    };

    bf16* q_bf    = alloc_bf((size_t)NTQ * QDIM);
    bf16* kv_bf   = alloc_bf((size_t)NTK * KVDIM);
    bf16* wkvcT   = alloc_bf((size_t)KVDIM * KVC);      // [256][1024]
    bf16* wqdT    = alloc_bf((size_t)QDIM * QC);        // [384][1024]
    bf16* wquT    = alloc_bf((size_t)QC * H * HD);      // [1024][384]
    bf16* wqrT    = alloc_bf((size_t)QC * H * RR);      // [1024][384]
    bf16* wkuT    = alloc_bf((size_t)KVC * H * HD);     // [1024][256]
    bf16* wkrT    = alloc_bf((size_t)KVC * RR);         // [64][256]
    bf16* wvuT    = alloc_bf((size_t)KVC * H * HD);     // [1024][256]
    bf16* woT     = alloc_bf((size_t)(H * HD) * QDIM);  // [1024][1024]
    bf16* qc_bf   = alloc_bf((size_t)NTQ * QC);
    bf16* kvc_bf  = alloc_bf((size_t)NTK * KVC);
    bf16* qn_bf   = alloc_bf((size_t)NTQ * H * HD);
    bf16* qr_bf   = alloc_bf((size_t)NTQ * H * RR);
    bf16* kn_bf   = alloc_bf((size_t)NTK * H * HD);
    bf16* kr_bf   = alloc_bf((size_t)NTK * RR);
    bf16* v_bf    = alloc_bf((size_t)NTK * H * HD);
    bf16* ao_bf   = alloc_bf((size_t)NTQ * H * HD);
    (void)ws_size; (void)in_sizes; (void)n_in; (void)out_size;

    // 1) fp32 -> bf16 conversions (activations plain; weights transposed)
    launch_cvt (q,        q_bf,  (long)NTQ * QDIM,  stream);
    launch_cvt (kv,       kv_bf, (long)NTK * KVDIM, stream);
    launch_cvtT(w_kvcomp, wkvcT, KVDIM, KVC,    stream);
    launch_cvtT(w_qdown,  wqdT,  QDIM,  QC,     stream);
    launch_cvtT(w_qup,    wquT,  QC,    H * HD, stream);
    launch_cvtT(w_qrope,  wqrT,  QC,    H * RR, stream);
    launch_cvtT(w_kup,    wkuT,  KVC,   H * HD, stream);
    launch_cvtT(w_krope,  wkrT,  KVC,   RR,     stream);
    launch_cvtT(w_vup,    wvuT,  KVC,   H * HD, stream);
    launch_cvtT(w_out,    woT,   H * HD, QDIM,  stream);

    // 2) down-projections
    gemm_bf16out(q_bf,  wqdT,  qc_bf,  NTQ, QC,  QDIM,  stream);
    gemm_bf16out(kv_bf, wkvcT, kvc_bf, NTK, KVC, KVDIM, stream);

    // 3) up-projections
    gemm_bf16out(qc_bf,  wquT, qn_bf, NTQ, H * HD, QC,  stream);
    gemm_bf16out(qc_bf,  wqrT, qr_bf, NTQ, H * RR, QC,  stream);
    gemm_bf16out(kvc_bf, wkuT, kn_bf, NTK, H * HD, KVC, stream);
    gemm_bf16out(kvc_bf, wkrT, kr_bf, NTK, RR,     KVC, stream);
    gemm_bf16out(kvc_bf, wvuT, v_bf,  NTK, H * HD, KVC, stream);

    // 4) RoPE
    {
        long n = (long)NTQ * H * 32;
        rope_q_kernel<<<(unsigned)((n + 255) / 256), 256, 0, stream>>>(qr_bf);
        long m = (long)NTK * 32;
        rope_k_kernel<<<(unsigned)((m + 255) / 256), 256, 0, stream>>>(kr_bf);
    }

    // 5) sparse attention: 8 waves/block, one wave per (token, head)
    {
        unsigned blocks = (unsigned)((long)NTQ * H / 8);
        attn_sparse<<<blocks, 256, 0, stream>>>(qn_bf, qr_bf, kn_bf, kr_bf, v_bf,
                                                seg_id, ao_bf);
    }

    // 6) output projection, fp32 accumulators straight into d_out
    gemm_f32out(ao_bf, woT, (float*)d_out, NTQ, QDIM, H * HD, stream);
}
